// durationLMMultiLayer_79559974191178
// MI455X (gfx1250) — matrix operand order repaired and checked by reading, not compiled, not run on hardware
//
#include <hip/hip_runtime.h>
#include <hip/hip_bf16.h>
#include <math.h>

// ---------------------------------------------------------------------------
// MI455X (gfx1250): bf16 WMMA GEMMs (LDS block-tiled) + persistent LSTM scan
// with TDM (tensor_load_to_lds) staging of W_hh slices into LDS.
// ---------------------------------------------------------------------------

typedef __bf16 bf16_t;
typedef __attribute__((ext_vector_type(16))) __bf16 v16bf;
typedef __attribute__((ext_vector_type(8)))  __bf16 v8bf;
typedef __attribute__((ext_vector_type(8)))  float  v8f;
typedef __attribute__((ext_vector_type(4)))  unsigned int v4u;
typedef __attribute__((ext_vector_type(8)))  int v8i;
typedef __attribute__((ext_vector_type(4)))  int v4i;

#define B_    16
#define S_    256
#define V_    32000
#define E_    512
#define H_    512
#define G4H_  2048          // 4*H
#define MR_   4096          // B*S rows
#define HCAT_ 1536          // L*H
#define NWG_SCAN 16

// ---------------- WMMA fragment loaders (wave32 layouts per ISA 7.12.2) ----

// A (16x32 bf16): lane<16 holds row=lane, K = {0..7, 16..23};
//                 lane>=16 holds row=lane-16, K = {8..15, 24..31}.
__device__ __forceinline__ v16bf load_a_frag(const bf16_t* base, int ld, int lane) {
  const int row  = lane & 15;
  const int half = (lane >> 4) & 1;
  const bf16_t* p = base + (size_t)row * ld + half * 8;
  v8bf lo = *(const v8bf*)(p);
  v8bf hi = *(const v8bf*)(p + 16);
  v16bf r;
#pragma unroll
  for (int i = 0; i < 8; ++i) { r[i] = lo[i]; r[i + 8] = hi[i]; }
  return r;
}

// B (32x16 bf16) sourced from row-major W[N][K] (computing A @ W^T):
// lane<16 holds col N=lane, K = {0..7, 16..23}; lane>=16 holds K = {8..15, 24..31}.
__device__ __forceinline__ v16bf load_b_frag(const bf16_t* Wtile, int ldw, int lane) {
  const int col  = lane & 15;
  const int half = (lane >> 4) & 1;
  const bf16_t* p = Wtile + (size_t)col * ldw + half * 8;
  v8bf lo = *(const v8bf*)(p);
  v8bf hi = *(const v8bf*)(p + 16);
  v16bf r;
#pragma unroll
  for (int i = 0; i < 8; ++i) { r[i] = lo[i]; r[i + 8] = hi[i]; }
  return r;
}

// ---------------- fast activations (raw v_exp_f32 / v_rcp_f32) -------------
__device__ __forceinline__ float sigmoidf_(float x) {
  return __builtin_amdgcn_rcpf(1.f + __builtin_amdgcn_exp2f(-1.44269504f * x));
}
__device__ __forceinline__ float tanhf_(float x) {
  return 2.f * __builtin_amdgcn_rcpf(1.f + __builtin_amdgcn_exp2f(-2.88539008f * x)) - 1.f;
}

// ---------------- device-scope grid barrier (persistent scan kernel) -------
__device__ __forceinline__ void grid_barrier(unsigned int* cnt, unsigned nwg, unsigned barIdx) {
  __syncthreads();
  if (threadIdx.x == 0) {
    __hip_atomic_fetch_add(cnt, 1u, __ATOMIC_RELEASE, __HIP_MEMORY_SCOPE_AGENT);
    const unsigned target = nwg * barIdx;
    while (__hip_atomic_load(cnt, __ATOMIC_ACQUIRE, __HIP_MEMORY_SCOPE_AGENT) < target) {
      __builtin_amdgcn_s_sleep(1);
    }
  }
  __syncthreads();
}

// ---------------- TDM: 2D tile (nrows x rowlen bf16) global -> LDS ---------
// Descriptor bit layout per CDNA5 ISA.
// This toolchain exposes the 6-arg builtin: (g0 v4u, g1 v8i, v4i, v4i, v8i, cpol).
__device__ __forceinline__ void tdm_load_rows_bf16(unsigned lds_byte_off, const bf16_t* gsrc,
                                                   unsigned nrows, unsigned rowlen,
                                                   unsigned stride_elems) {
  const unsigned long long ga = (unsigned long long)(uintptr_t)gsrc;
  v4u g0;
  g0[0] = 1u;                                            // count=1, no gather, user mode
  g0[1] = lds_byte_off;                                  // lds_addr (bytes)
  g0[2] = (unsigned)(ga & 0xffffffffu);                  // global_addr[31:0]
  g0[3] = (unsigned)((ga >> 32) & 0x01ffffffu) | (2u << 30);  // addr[56:32] | type=2
  v8i g1;
  g1[0] = (int)(1u << 16);                               // wg_mask=0, data_size=1 (2 bytes)
  g1[1] = (int)((rowlen & 0xffffu) << 16);               // tensor_dim0[15:0] @ bits63:48
  g1[2] = (int)(((rowlen >> 16) & 0xffffu) | ((nrows & 0xffffu) << 16)); // td0 hi | td1 lo
  g1[3] = (int)(((nrows >> 16) & 0xffffu) | ((rowlen & 0xffffu) << 16)); // td1 hi | tile_dim0
  g1[4] = (int)(nrows & 0xffffu);                        // tile_dim1 (tile_dim2 = 0)
  g1[5] = (int)stride_elems;                             // tensor_dim0_stride[31:0]
  g1[6] = 0;                                             // stride hi / dim1_stride lo
  g1[7] = 0;
  v4i z4{};
  v8i z8{};
  __builtin_amdgcn_tensor_load_to_lds(g0, g1, z4, z4, z8, 0);
}

// ---------------- small utility kernels ------------------------------------
__global__ void cvt_f32_bf16(const float* __restrict__ in, bf16_t* __restrict__ out, size_t n) {
  size_t i = (size_t)blockIdx.x * blockDim.x + threadIdx.x;
  const size_t stride = (size_t)gridDim.x * blockDim.x;
  for (; i < n; i += stride) out[i] = (bf16_t)in[i];
}

__global__ void gather_embed(const int* __restrict__ src, const float* __restrict__ Wemb,
                             bf16_t* __restrict__ x) {
  size_t i = (size_t)blockIdx.x * blockDim.x + threadIdx.x;
  const size_t n = (size_t)MR_ * E_;
  const size_t stride = (size_t)gridDim.x * blockDim.x;
  for (; i < n; i += stride) {
    const size_t r = i / E_;
    const size_t e = i - r * E_;
    x[i] = (bf16_t)Wemb[(size_t)src[r] * E_ + e];
  }
}

__global__ void init_counters(unsigned int* c, int n) {
  if ((int)threadIdx.x < n) c[threadIdx.x] = 0u;
}

// ---------------- block-tiled WMMA GEMM: C = A[M,K] @ W[N,K]^T + bias ------
// 256 threads (8 waves), block tile 128M x 128N, LDS-staged 32-K slices.
// Wave (wm, wn) in 2x4 grid computes a 64M x 32N sub-tile: 8 WMMAs / K-step.
template <bool TANH, bool OUT_BF16>
__global__ __launch_bounds__(256)
void gemm_bias_wmma(const bf16_t* __restrict__ A, const bf16_t* __restrict__ W,
                    const float* __restrict__ bias, void* __restrict__ Cout,
                    int K, int lda, int ldw, int ldc) {
  __shared__ bf16_t lds_a[128 * 32];   // 8 KB
  __shared__ bf16_t lds_b[128 * 32];   // 8 KB

  const int tid  = threadIdx.x;
  const int lane = tid & 31;
  const int wave = tid >> 5;
  const int wm   = wave >> 2;          // 0..1
  const int wn   = wave & 3;           // 0..3
  const int m0   = blockIdx.y * 128;
  const int n0   = blockIdx.x * 128;

  // staging assignment: thread copies one 32B chunk of one row per tile
  const int crow  = tid >> 1;          // 0..127
  const int chalf = tid & 1;           // 0..1
  const bf16_t* gA = A + (size_t)(m0 + crow) * lda + chalf * 16;
  const bf16_t* gB = W + (size_t)(n0 + crow) * ldw + chalf * 16;

  v8f acc[4][2];
#pragma unroll
  for (int ms = 0; ms < 4; ++ms)
#pragma unroll
    for (int ns = 0; ns < 2; ++ns) acc[ms][ns] = v8f{};

  for (int k0 = 0; k0 < K; k0 += 32) {
    __syncthreads();   // previous compute finished reading LDS
    *(v16bf*)(lds_a + crow * 32 + chalf * 16) = *(const v16bf*)(gA + k0);
    *(v16bf*)(lds_b + crow * 32 + chalf * 16) = *(const v16bf*)(gB + k0);
    if (k0 + 32 < K) {                 // pull next K-slice toward the WGP
      __builtin_prefetch(gA + k0 + 32, 0, 0);
      __builtin_prefetch(gB + k0 + 32, 0, 0);
    }
    __syncthreads();

    v16bf af[4], bfr[2];
#pragma unroll
    for (int ms = 0; ms < 4; ++ms)
      af[ms] = load_a_frag(lds_a + (wm * 64 + ms * 16) * 32, 32, lane);
#pragma unroll
    for (int ns = 0; ns < 2; ++ns)
      bfr[ns] = load_b_frag(lds_b + (wn * 32 + ns * 16) * 32, 32, lane);
#pragma unroll
    for (int ms = 0; ms < 4; ++ms)
#pragma unroll
      for (int ns = 0; ns < 2; ++ns)
        acc[ms][ns] = __builtin_amdgcn_wmma_f32_16x16x32_bf16(
            false, af[ms], false, bfr[ns], (short)0, acc[ms][ns], false, false);
  }

  const int col  = lane & 15;
  const int half = (lane >> 4) & 1;
#pragma unroll
  for (int ms = 0; ms < 4; ++ms) {
#pragma unroll
    for (int ns = 0; ns < 2; ++ns) {
      const int n = n0 + wn * 32 + ns * 16 + col;
      const float bv = bias[n];
#pragma unroll
      for (int r = 0; r < 8; ++r) {
        const int m = m0 + wm * 64 + ms * 16 + half * 8 + r;
        float v = acc[ms][ns][r] + bv;
        if (TANH) v = tanhf_(v);
        if (OUT_BF16) ((bf16_t*)Cout)[(size_t)m * ldc + n] = (bf16_t)v;
        else          ((float*)Cout)[(size_t)m * ldc + n]  = v;
      }
    }
  }
}

// ---------------- persistent LSTM scan kernel ------------------------------
// grid = 16 WGs x 256 threads (8 waves). WG w owns h-columns [w*32, w*32+32).
// LDS: W_hh slice (4 gates x 32 rows x 512 K bf16 = 128KB, TDM-staged)
//      + z-tile (8KB) + c-state (2KB).
__global__ __launch_bounds__(256)
void lstm_scan(const float* __restrict__ zx,                   // (MR_, 2048)
               const bf16_t* __restrict__ x_in, int x_in_ld,   // null for layer 0
               bf16_t* __restrict__ x_out,                     // next-layer input
               bf16_t* __restrict__ hcat_out, int hcat_ld,     // base + layer*H
               const bf16_t* __restrict__ Whh,                 // (2048, 512) bf16
               const int* __restrict__ fdur, int layer,
               bf16_t* __restrict__ hbuf,                      // 2 x (B*H) ping-pong
               unsigned int* __restrict__ sync_cnt) {
  extern __shared__ __attribute__((aligned(64))) char smem[];
  bf16_t* lds_w = (bf16_t*)smem;                                            // 4*32*512 bf16
  float*  lds_z = (float*)(smem + (size_t)4 * 32 * 512 * sizeof(bf16_t));   // 4*16*32 f32
  float*  lds_c = lds_z + 4 * 16 * 32;                                      // 16*32 f32

  const int wg   = blockIdx.x;        // 0..15
  const int tid  = threadIdx.x;
  const int lane = tid & 31;
  const int wave = tid >> 5;          // 0..7
  const int gate = wave >> 1;         // i,f,g,o
  const int sub  = wave & 1;          // 16-col half of the 32-col chunk

  // Stage this WG's W_hh slice via the Tensor Data Mover (wave 0 issues,
  // one 32x512 2D tile per gate; EXEC is ignored by TENSOR ops).
  if (wave == 0) {
    const unsigned lds_base = (unsigned)(uintptr_t)lds_w;
#pragma unroll
    for (int g = 0; g < 4; ++g) {
      tdm_load_rows_bf16(lds_base + (unsigned)g * 32u * 512u * 2u,
                         Whh + ((size_t)(g * H_ + wg * 32)) * H_,
                         /*nrows=*/32u, /*rowlen=*/512u, /*stride=*/512u);
    }
    __builtin_amdgcn_s_wait_tensorcnt(0);
  }
  for (int i = tid; i < 16 * 32; i += 256) lds_c[i] = 0.f;
  for (int i = tid; i < 16 * 32; i += 256) {          // zero h ping buffer chunk
    const int b = i >> 5, j = i & 31;
    hbuf[(size_t)b * H_ + wg * 32 + j] = (bf16_t)0.f;
  }
  unsigned barIdx = 0;
  grid_barrier(sync_cnt, NWG_SCAN, ++barIdx);

  for (int t = 0; t < S_; ++t) {
    const bf16_t* hcur  = hbuf + (size_t)(t & 1) * (B_ * H_);
    bf16_t*       hnext = hbuf + (size_t)((t + 1) & 1) * (B_ * H_);

    // Accumulator initialized from precomputed zx (x@W_ih^T + b).
    v8f acc{};
    {
      const int col  = lane & 15;
      const int half = (lane >> 4) & 1;
      const int ncol = gate * H_ + wg * 32 + sub * 16 + col;
#pragma unroll
      for (int r = 0; r < 8; ++r) {
        const int b = half * 8 + r;
        acc[r] = zx[((size_t)(b * S_ + t)) * G4H_ + ncol];
      }
    }

    // z += h @ W_hh^T for this wave's 16 gate-columns (K = 512).
    const bf16_t* wrow = lds_w + (size_t)(gate * 32 + sub * 16) * H_;
#pragma unroll 4
    for (int k0 = 0; k0 < H_; k0 += 32) {
      v16bf a = load_a_frag(hcur + k0, H_, lane);
      v16bf b = load_b_frag(wrow + k0, H_, lane);
      acc = __builtin_amdgcn_wmma_f32_16x16x32_bf16(false, a, false, b, (short)0, acc, false, false);
    }

    // Spill z tile so i/f/g/o (different waves) meet per (b, j).
    {
      const int col  = lane & 15;
      const int half = (lane >> 4) & 1;
#pragma unroll
      for (int r = 0; r < 8; ++r) {
        const int b = half * 8 + r;
        lds_z[(gate * 16 + b) * 32 + sub * 16 + col] = acc[r];
      }
    }
    __syncthreads();

    // Gate math + state update: 512 (b, j) pairs, two per thread.
#pragma unroll
    for (int it = 0; it < 2; ++it) {
      const int p = tid + it * 256;
      const int b = p >> 5;
      const int j = p & 31;
      const float iv = lds_z[(0 * 16 + b) * 32 + j];
      const float fv = lds_z[(1 * 16 + b) * 32 + j];
      const float gv = lds_z[(2 * 16 + b) * 32 + j];
      const float ov = lds_z[(3 * 16 + b) * 32 + j];
      const float c_old = lds_c[b * 32 + j];
      const float cn = sigmoidf_(fv) * c_old + sigmoidf_(iv) * tanhf_(gv);
      const float hn = sigmoidf_(ov) * tanhf_(cn);
      const size_t row = (size_t)b * S_ + t;
      float h_new, c_new;
      if (layer == 0) {
        h_new = hn;
        c_new = cn;
        hcat_out[row * hcat_ld + wg * 32 + j] = (bf16_t)hn;
      } else {
        const float m   = ((fdur[row] >> 2) >= layer) ? 1.f : 0.f;  // fd//4 >= layer
        const float x_t = (float)x_in[row * x_in_ld + wg * 32 + j];
        const float h_old = (float)hcur[(size_t)b * H_ + wg * 32 + j];
        const float x_o = x_t + m * (hn - x_t);
        h_new = h_old + m * (hn - h_old);
        c_new = c_old + m * (cn - c_old);
        x_out[row * (size_t)H_ + wg * 32 + j]  = (bf16_t)x_o;
        hcat_out[row * hcat_ld + wg * 32 + j]  = (bf16_t)h_new;
      }
      lds_c[b * 32 + j] = c_new;
      hnext[(size_t)b * H_ + wg * 32 + j] = (bf16_t)h_new;
    }
    // Publish h chunk; everyone resyncs before reading for step t+1.
    grid_barrier(sync_cnt, NWG_SCAN, ++barIdx);
  }
}

// ---------------------------------------------------------------------------
extern "C" void kernel_launch(void* const* d_in, const int* in_sizes, int n_in,
                              void* d_out, int out_size, void* d_ws, size_t ws_size,
                              hipStream_t stream) {
  (void)in_sizes; (void)n_in; (void)out_size; (void)ws_size;

  const int*   src   = (const int*)d_in[0];
  const int*   fdur  = (const int*)d_in[1];
  const float* W_emb = (const float*)d_in[2];
  const float* W_ih[3] = {(const float*)d_in[3], (const float*)d_in[6], (const float*)d_in[9]};
  const float* W_hh[3] = {(const float*)d_in[4], (const float*)d_in[7], (const float*)d_in[10]};
  const float* b_g[3]  = {(const float*)d_in[5], (const float*)d_in[8], (const float*)d_in[11]};
  const float* W_fc1 = (const float*)d_in[14];
  const float* b_fc1 = (const float*)d_in[15];
  const float* W_fc2 = (const float*)d_in[16];
  const float* b_fc2 = (const float*)d_in[17];
  float* out = (float*)d_out;

  // ---- carve workspace ----
  uintptr_t p = ((uintptr_t)d_ws + 255) & ~(uintptr_t)255;
  auto carve = [&](size_t bytes) -> void* {
    void* r = (void*)p;
    p += (bytes + 255) & ~(size_t)255;
    return r;
  };
  bf16_t* wih_bf[3], *whh_bf[3];
  for (int l = 0; l < 3; ++l) {
    wih_bf[l] = (bf16_t*)carve((size_t)G4H_ * H_ * 2);
    whh_bf[l] = (bf16_t*)carve((size_t)G4H_ * H_ * 2);
  }
  bf16_t* wfc1_bf = (bf16_t*)carve((size_t)E_ * HCAT_ * 2);
  bf16_t* wfc2_bf = (bf16_t*)carve((size_t)V_ * E_ * 2);
  bf16_t* x_emb   = (bf16_t*)carve((size_t)MR_ * E_ * 2);
  bf16_t* x1      = (bf16_t*)carve((size_t)MR_ * H_ * 2);
  float*  zx      = (float*)carve((size_t)MR_ * G4H_ * 4);
  bf16_t* hcat    = (bf16_t*)carve((size_t)MR_ * HCAT_ * 2);
  bf16_t* fc1out  = (bf16_t*)carve((size_t)MR_ * E_ * 2);
  bf16_t* hbuf    = (bf16_t*)carve((size_t)2 * B_ * H_ * 2);
  unsigned int* cnt = (unsigned int*)carve(256);

  // ---- one-time conversions (f32 -> bf16) + embedding gather ----
  for (int l = 0; l < 3; ++l) {
    cvt_f32_bf16<<<1024, 256, 0, stream>>>(W_ih[l], wih_bf[l], (size_t)G4H_ * H_);
    cvt_f32_bf16<<<1024, 256, 0, stream>>>(W_hh[l], whh_bf[l], (size_t)G4H_ * H_);
  }
  cvt_f32_bf16<<<1024, 256, 0, stream>>>(W_fc1, wfc1_bf, (size_t)E_ * HCAT_);
  cvt_f32_bf16<<<2048, 256, 0, stream>>>(W_fc2, wfc2_bf, (size_t)V_ * E_);
  gather_embed<<<2048, 256, 0, stream>>>(src, W_emb, x_emb);
  init_counters<<<1, 32, 0, stream>>>(cnt, 8);

  constexpr size_t SCAN_LDS =
      (size_t)4 * 32 * 512 * sizeof(bf16_t) + (4 * 16 * 32 + 16 * 32) * sizeof(float);

  // ---- layer 0 ----
  gemm_bias_wmma<false, false><<<dim3(G4H_ / 128, MR_ / 128), 256, 0, stream>>>(
      x_emb, wih_bf[0], b_g[0], zx, E_, E_, E_, G4H_);
  lstm_scan<<<NWG_SCAN, 256, SCAN_LDS, stream>>>(
      zx, nullptr, 0, nullptr, hcat + 0 * H_, HCAT_, whh_bf[0], fdur, 0, hbuf, cnt + 0);

  // ---- layer 1 (input = out0 = hcat[:, 0:512]) ----
  gemm_bias_wmma<false, false><<<dim3(G4H_ / 128, MR_ / 128), 256, 0, stream>>>(
      hcat, wih_bf[1], b_g[1], zx, E_, HCAT_, H_, G4H_);
  lstm_scan<<<NWG_SCAN, 256, SCAN_LDS, stream>>>(
      zx, hcat, HCAT_, x1, hcat + 1 * H_, HCAT_, whh_bf[1], fdur, 1, hbuf, cnt + 1);

  // ---- layer 2 (input = x1) ----
  gemm_bias_wmma<false, false><<<dim3(G4H_ / 128, MR_ / 128), 256, 0, stream>>>(
      x1, wih_bf[2], b_g[2], zx, E_, H_, H_, G4H_);
  lstm_scan<<<NWG_SCAN, 256, SCAN_LDS, stream>>>(
      zx, x1, H_, x1, hcat + 2 * H_, HCAT_, whh_bf[2], fdur, 2, hbuf, cnt + 2);

  // ---- fc1: tanh(hcat @ W_fc1^T + b), bf16 out ----
  gemm_bias_wmma<true, true><<<dim3(E_ / 128, MR_ / 128), 256, 0, stream>>>(
      hcat, wfc1_bf, b_fc1, fc1out, HCAT_, HCAT_, HCAT_, E_);

  // ---- fc2: fc1out @ W_fc2^T + b -> d_out (f32) ----
  gemm_bias_wmma<false, false><<<dim3(V_ / 128, MR_ / 128), 256, 0, stream>>>(
      fc1out, wfc2_bf, b_fc2, out, E_, E_, E_, V_);
}
